// ActorSlowInParallel_65816078844575
// MI455X (gfx1250) — compile-verified, matrix operand order checked by reading, not run on hardware
//
#include <hip/hip_runtime.h>
#include <hip/hip_bf16.h>
#include <math.h>

#define BATCH   8192
#define HID     512
#define OBS     21
#define ACT     8
#define NLAY    10
#define OVERALL (HID * (NLAY + 1))   // 5632
#define IN_DIM  (OBS + OVERALL)      // 5653
#define OUT_DIM (OVERALL + ACT)      // 5640

typedef __attribute__((ext_vector_type(16))) __bf16       v16bf;
typedef __attribute__((ext_vector_type(8)))  float        v8f;
typedef __attribute__((ext_vector_type(8)))  unsigned int v8u;

__device__ __forceinline__ v8f wmma_bf16(v16bf a, v16bf b, v8f c) {
    return __builtin_amdgcn_wmma_f32_16x16x32_bf16(false, a, false, b, (short)0, c,
                                                   false, false);
}

// A fragment: f32 source, 16B-aligned, row already applied by caller.
// Lane fragment = f32 p[0..7] -> halves 0..7, p[16..23] -> halves 8..15.
__device__ __forceinline__ v16bf load_a_f32(const float* __restrict__ p) {
    const float4* q = reinterpret_cast<const float4*>(p);
    float4 f0 = q[0], f1 = q[1], f2 = q[4], f3 = q[5];
    v16bf r;
    r[0]  = (__bf16)f0.x; r[1]  = (__bf16)f0.y; r[2]  = (__bf16)f0.z; r[3]  = (__bf16)f0.w;
    r[4]  = (__bf16)f1.x; r[5]  = (__bf16)f1.y; r[6]  = (__bf16)f1.z; r[7]  = (__bf16)f1.w;
    r[8]  = (__bf16)f2.x; r[9]  = (__bf16)f2.y; r[10] = (__bf16)f2.z; r[11] = (__bf16)f2.w;
    r[12] = (__bf16)f3.x; r[13] = (__bf16)f3.y; r[14] = (__bf16)f3.z; r[15] = (__bf16)f3.w;
    return r;
}

// B fragment: pre-converted bf16 source (16B-aligned, row stride HID elems).
// halves 0..7 = p[0..7] (16 bytes), halves 8..15 = p[16..23] (16 bytes at +32B).
__device__ __forceinline__ v16bf load_b_bf16(const __bf16* __restrict__ p) {
    const uint4* q = reinterpret_cast<const uint4*>(p);
    uint4 q0 = q[0];
    uint4 q1 = q[2];
    v8u u;
    u[0] = q0.x; u[1] = q0.y; u[2] = q0.z; u[3] = q0.w;
    u[4] = q1.x; u[5] = q1.y; u[6] = q1.z; u[7] = q1.w;
    return __builtin_bit_cast(v16bf, u);
}

// ---- Kernel 0: one-time (per launch) bf16 conversion of the ten banded
// 512x512 W_mean blocks into d_ws, dense layout Wb[j-1][n][k], k-stride 1.
__global__ __launch_bounds__(256) void wprep_kernel(
    const float* __restrict__ W_mean, __bf16* __restrict__ Wb) {
    const int idx  = blockIdx.x * 256 + threadIdx.x;  // 10*512*256 threads
    const int kp   = idx & 255;                       // k pair index
    const int rest = idx >> 8;
    const int n    = rest & 511;
    const int jm1  = rest >> 9;                       // 0..9
    const float* src = W_mean + (size_t)(HID * (jm1 + 1) + n) * IN_DIM + OBS +
                       (size_t)jm1 * HID + 2 * kp;
    const __bf16 h0 = (__bf16)src[0];
    const __bf16 h1 = (__bf16)src[1];
    const unsigned u = (unsigned)__builtin_bit_cast(unsigned short, h0) |
                       ((unsigned)__builtin_bit_cast(unsigned short, h1) << 16);
    *reinterpret_cast<unsigned*>(Wb + (size_t)jm1 * HID * HID + (size_t)n * HID +
                                 2 * kp) = u;
}

// ---- Kernel 1: the ten 8192x512x512 banded blocks via bf16 WMMA ----------
// grid = (BATCH/128, HID/128, NLAY); block = 256 (8 waves).
// Wave computes a 32x64 output tile (2 M-frags x 4 N-frags -> 8 WMMA / K-step).
__global__ __launch_bounds__(256) void gemm_mid_kernel(
    const float* __restrict__ hidden0, const __bf16* __restrict__ Wb,
    const float* __restrict__ b_mean, float* __restrict__ out_hidden) {
    const int j    = blockIdx.z + 1;           // banded block 1..10
    const int m0   = blockIdx.x * 128;
    const int n0   = blockIdx.y * 128;
    const int wave = threadIdx.x >> 5;
    const int lane = threadIdx.x & 31;
    const int wm   = wave >> 1;                // 0..3 -> M
    const int wn   = wave & 1;                 // 0..1 -> N
    const int row0 = m0 + wm * 32;
    const int col0 = n0 + wn * 64;
    const int sub  = lane >> 4;
    const int r    = lane & 15;

    const float* A0 = hidden0 + (size_t)(row0 + r) * OVERALL +
                      (size_t)(j - 1) * HID + sub * 8;
    const float* A1 = A0 + (size_t)16 * OVERALL;
    const __bf16* Bb = Wb + (size_t)(j - 1) * HID * HID +
                       (size_t)(col0 + r) * HID + sub * 8;

    v8f acc[2][4];
#pragma unroll
    for (int i = 0; i < 2; ++i)
#pragma unroll
        for (int t = 0; t < 4; ++t) acc[i][t] = (v8f){};

    for (int k0 = 0; k0 < HID; k0 += 32) {
        v16bf a0 = load_a_f32(A0 + k0);
        v16bf a1 = load_a_f32(A1 + k0);
        v16bf b[4];
#pragma unroll
        for (int t = 0; t < 4; ++t) b[t] = load_b_bf16(Bb + (size_t)t * 16 * HID + k0);
#pragma unroll
        for (int t = 0; t < 4; ++t) {
            acc[0][t] = wmma_bf16(a0, b[t], acc[0][t]);
            acc[1][t] = wmma_bf16(a1, b[t], acc[1][t]);
        }
    }

    // Epilogue: C/D layout -> N = lane&15, M = (lane>=16 ? 8:0) + vgpr
    const int n  = lane & 15;
    const int mh = (lane >> 4) * 8;
#pragma unroll
    for (int t = 0; t < 4; ++t) {
        const int   col  = HID * j + col0 + t * 16 + n;  // column in new_hidden
        const float bias = b_mean[col];
#pragma unroll
        for (int v = 0; v < 8; ++v) {
            const int m_a = row0 + mh + v;
            const int m_b = row0 + 16 + mh + v;
            out_hidden[(size_t)m_a * OVERALL + col] = fmaxf(acc[0][t][v] + bias, 0.0f);
            out_hidden[(size_t)m_b * OVERALL + col] = fmaxf(acc[1][t][v] + bias, 0.0f);
        }
    }
}

// ---- Kernel 2: block 0 (K = 21, too small for WMMA) ----------------------
__global__ __launch_bounds__(256) void block0_kernel(
    const float* __restrict__ obs, const float* __restrict__ W_mean,
    const float* __restrict__ b_mean, float* __restrict__ out_hidden) {
    const int idx = blockIdx.x * 256 + threadIdx.x;  // BATCH*HID threads
    const int b   = idx >> 9;
    const int n   = idx & 511;
    const float* o = obs + (size_t)b * OBS;
    const float* w = W_mean + (size_t)n * IN_DIM;
    float acc = b_mean[n];
#pragma unroll
    for (int k = 0; k < OBS; ++k) acc = fmaf(o[k], w[k], acc);
    out_hidden[(size_t)b * OVERALL + n] = fmaxf(acc, 0.0f);
}

// ---- Kernel 3: block 11 -> new_mean (8 outputs, K = 512, no relu) --------
__global__ __launch_bounds__(256) void mean_tail_kernel(
    const float* __restrict__ hidden0, const float* __restrict__ W_mean,
    const float* __restrict__ b_mean, float* __restrict__ out_new_mean) {
    const int wid  = (blockIdx.x * blockDim.x + threadIdx.x) >> 5;
    const int lane = threadIdx.x & 31;
    if (wid >= BATCH) return;
    const float* xh = hidden0 + (size_t)wid * OVERALL + (size_t)NLAY * HID;
    float acc[ACT];
#pragma unroll
    for (int a = 0; a < ACT; ++a) acc[a] = 0.0f;
    for (int k = lane; k < HID; k += 32) {
        const float xv = xh[k];
#pragma unroll
        for (int a = 0; a < ACT; ++a)
            acc[a] = fmaf(xv,
                          W_mean[(size_t)(OVERALL + a) * IN_DIM + OBS +
                                 (size_t)NLAY * HID + k],
                          acc[a]);
    }
#pragma unroll
    for (int a = 0; a < ACT; ++a)
#pragma unroll
        for (int m = 16; m >= 1; m >>= 1) acc[a] += __shfl_xor(acc[a], m, 32);
    if (lane == 0) {
#pragma unroll
        for (int a = 0; a < ACT; ++a)
            out_new_mean[(size_t)wid * ACT + a] = acc[a] + b_mean[OVERALL + a];
    }
}

// ---- Kernel 4: prev_mean copy, log_std transform, new_log_std GEMM -------
__global__ __launch_bounds__(256) void head_kernel(
    const float* __restrict__ obs, const float* __restrict__ hidden0,
    const float* __restrict__ prev_mean, const float* __restrict__ prev_logstd,
    const float* __restrict__ W_logstd, const float* __restrict__ b_logstd,
    float* __restrict__ out_prev_mean, float* __restrict__ out_log_std,
    float* __restrict__ out_new_log_std) {
    const int wid  = (blockIdx.x * blockDim.x + threadIdx.x) >> 5;
    const int lane = threadIdx.x & 31;
    if (wid >= BATCH) return;
    const float* xo = obs + (size_t)wid * OBS;
    const float* xh = hidden0 + (size_t)wid * OVERALL;
    float acc[ACT];
#pragma unroll
    for (int a = 0; a < ACT; ++a) acc[a] = 0.0f;
    if (lane < OBS) {
        const float xv = xo[lane];
#pragma unroll
        for (int a = 0; a < ACT; ++a)
            acc[a] = fmaf(xv, W_logstd[(size_t)a * IN_DIM + lane], acc[a]);
    }
    for (int k = lane; k < OVERALL; k += 32) {
        const float xv = xh[k];
#pragma unroll
        for (int a = 0; a < ACT; ++a)
            acc[a] = fmaf(xv, W_logstd[(size_t)a * IN_DIM + OBS + k], acc[a]);
    }
#pragma unroll
    for (int a = 0; a < ACT; ++a)
#pragma unroll
        for (int m = 16; m >= 1; m >>= 1) acc[a] += __shfl_xor(acc[a], m, 32);
    if (lane == 0) {
#pragma unroll
        for (int a = 0; a < ACT; ++a)
            out_new_log_std[(size_t)wid * ACT + a] = acc[a] + b_logstd[a];
    }
    if (lane < ACT) {
        out_prev_mean[(size_t)wid * ACT + lane] = prev_mean[(size_t)wid * ACT + lane];
        const float t = tanhf(prev_logstd[(size_t)wid * ACT + lane]);
        // LOG_STD_MIN + 0.5*(LOG_STD_MAX-LOG_STD_MIN)*(t+1) = -5 + 3.5*(t+1)
        out_log_std[(size_t)wid * ACT + lane] = -5.0f + 3.5f * (t + 1.0f);
    }
}

extern "C" void kernel_launch(void* const* d_in, const int* in_sizes, int n_in,
                              void* d_out, int out_size, void* d_ws, size_t ws_size,
                              hipStream_t stream) {
    const float* obs         = (const float*)d_in[0];
    const float* hidden0     = (const float*)d_in[1];
    const float* prev_mean   = (const float*)d_in[2];
    const float* prev_logstd = (const float*)d_in[3];
    const float* W_mean      = (const float*)d_in[4];
    const float* b_mean      = (const float*)d_in[5];
    const float* W_logstd    = (const float*)d_in[6];
    const float* b_logstd    = (const float*)d_in[7];

    float* out             = (float*)d_out;
    float* out_prev_mean   = out;                                       // [8192, 8]
    float* out_log_std     = out + (size_t)BATCH * ACT;                 // [8192, 8]
    float* out_new_hidden  = out + (size_t)2 * BATCH * ACT;             // [8192, 5632]
    float* out_new_mean    = out_new_hidden + (size_t)BATCH * OVERALL;  // [8192, 8]
    float* out_new_log_std = out_new_mean + (size_t)BATCH * ACT;        // [8192, 8]

    __bf16* Wb = (__bf16*)d_ws;  // 10*512*512 bf16 = 5.24 MB

    // Convert banded W blocks to bf16 (tiny; L2-resident afterwards)
    wprep_kernel<<<(NLAY * HID * (HID / 2)) / 256, 256, 0, stream>>>(W_mean, Wb);

    // Banded WMMA GEMM: blocks 1..10 of W_mean
    dim3 g(BATCH / 128, HID / 128, NLAY);
    gemm_mid_kernel<<<g, 256, 0, stream>>>(hidden0, Wb, b_mean, out_new_hidden);

    // Block 0 (obs -> first 512 hidden cols)
    block0_kernel<<<(BATCH * HID) / 256, 256, 0, stream>>>(obs, W_mean, b_mean,
                                                           out_new_hidden);

    // Block 11 (last hidden slab -> new_mean)
    mean_tail_kernel<<<BATCH / 8, 256, 0, stream>>>(hidden0, W_mean, b_mean,
                                                    out_new_mean);

    // prev_mean copy, log_std transform, new_log_std skinny GEMM
    head_kernel<<<BATCH / 8, 256, 0, stream>>>(obs, hidden0, prev_mean, prev_logstd,
                                               W_logstd, b_logstd, out_prev_mean,
                                               out_log_std, out_new_log_std);
}
